// MQGoraTransformerV4_52235392253993
// MI455X (gfx1250) — matrix operand order repaired and checked by reading, not compiled, not run on hardware
//
#include <hip/hip_runtime.h>

// ---------------------------------------------------------------------------
// Shapes (compile-time constants from the reference)
// B=4096, N=64, M=4, F=128, OBS=64, D=64, H=4, L=2, FF=128, HID=64, OUT=8
// ---------------------------------------------------------------------------

typedef __attribute__((ext_vector_type(16))) __bf16 bf16x16;
typedef __attribute__((ext_vector_type(8)))  float  f32x8;
typedef __attribute__((ext_vector_type(4)))  unsigned int u32x4;

union FragAB {
  bf16x16 v;
  u32x4   q[2];
};

__device__ __attribute__((always_inline)) inline
f32x8 wmma_bf16(const FragAB& a, const FragAB& b, f32x8 c) {
  // v_wmma_f32_16x16x32_bf16 : D = A(16x32) * B(32x16) + C(16x16 f32)
  return __builtin_amdgcn_wmma_f32_16x16x32_bf16(
      false, a.v, false, b.v, (short)0, c, false, false);
}

__device__ inline float gelu_exact(float x) {
  return 0.5f * x * (1.0f + erff(x * 0.70710678118654752f));
}

// ---------------------------------------------------------------------------
// 64x64 GEMM: C(64x64,f32,LDS) = A(64xK bf16, LDS, row-major, stride K)
//                              * B(Kx64 bf16)  given as BT (64xK row-major)
// 4 waves: wave w owns output rows [16w,16w+16); each wave computes 4 tiles.
// Caller must __syncthreads() before (A,BT ready) and after (C ready).
// ---------------------------------------------------------------------------
template <bool ADD_BIAS>
__device__ __attribute__((always_inline)) inline
void gemm64(const __bf16* __restrict__ A, int K,
            const __bf16* __restrict__ BT,
            float* __restrict__ C,
            const float* __restrict__ bias) {
  const int tid  = threadIdx.x;
  const int wave = tid >> 5;
  const int lane = tid & 31;
  const int lo   = lane & 15;
  const int hi   = lane >> 4;

  f32x8 acc[4] = {{0.f,0.f,0.f,0.f,0.f,0.f,0.f,0.f},
                  {0.f,0.f,0.f,0.f,0.f,0.f,0.f,0.f},
                  {0.f,0.f,0.f,0.f,0.f,0.f,0.f,0.f},
                  {0.f,0.f,0.f,0.f,0.f,0.f,0.f,0.f}};

  for (int k0 = 0; k0 < K; k0 += 32) {
    // A fragment: lane lo -> row (16w+lo); K = hi*8+{0..7} then +16
    FragAB a;
    const __bf16* ap = A + (wave * 16 + lo) * K + k0 + hi * 8;
    a.q[0] = *(const u32x4*)(ap);
    a.q[1] = *(const u32x4*)(ap + 16);
#pragma unroll
    for (int tj = 0; tj < 4; ++tj) {
      // B fragment from BT (N x K row-major): lane lo -> col (16tj+lo),
      // K = hi*8+{0..7} then +16
      FragAB b;
      const __bf16* bp = BT + (tj * 16 + lo) * K + k0 + hi * 8;
      b.q[0] = *(const u32x4*)(bp);
      b.q[1] = *(const u32x4*)(bp + 16);
      acc[tj] = wmma_bf16(a, b, acc[tj]);
    }
  }
#pragma unroll
  for (int tj = 0; tj < 4; ++tj) {
    const int n = tj * 16 + lo;
    const float bv = ADD_BIAS ? bias[n] : 0.0f;
#pragma unroll
    for (int r = 0; r < 8; ++r) {
      const int m = wave * 16 + r + 8 * hi;   // C/D layout: VGPR r -> row m
      C[m * 64 + n] = acc[tj][r] + bv;
    }
  }
}

// ---------------------------------------------------------------------------
// Kernel 0: convert f32 (R x C) -> bf16 transposed (C x R), batched
// ---------------------------------------------------------------------------
__global__ void k_convT(const float* __restrict__ src, __bf16* __restrict__ dst,
                        int R, int C) {
  const size_t base = (size_t)blockIdx.y * R * C;
  for (int i = blockIdx.x * blockDim.x + threadIdx.x; i < R * C;
       i += gridDim.x * blockDim.x) {
    const int r = i / C, c = i % C;
    dst[base + (size_t)c * R + r] = (__bf16)src[base + i];
  }
}

// ---------------------------------------------------------------------------
// Kernel 1: per-b neighbor embedding + per-view context
// ---------------------------------------------------------------------------
__global__ void __launch_bounds__(128) k_neighbor(
    const float* __restrict__ x_nei, const float* __restrict__ ew_anc,
    const float* __restrict__ bv_emb, const float* __restrict__ Wv_val,
    const float* __restrict__ bv_val, const __bf16* __restrict__ WvembT,
    __bf16* __restrict__ hns_ws, float* __restrict__ pvc_ws,
    float* __restrict__ ctxvec_ws) {
  __shared__ __align__(16) __bf16 Xn[64 * 128];
  __shared__ __align__(16) float  HM[64 * 64];
  __shared__ float wL[4 * 64];
  __shared__ float swn[4];
  __shared__ float cbar[64];
  __shared__ float ctxacc[64];

  const int b = blockIdx.x;
  const int t = threadIdx.x;

  const float* xn = x_nei + (size_t)b * 64 * 128;
  for (int i = t; i < 64 * 128; i += 128) Xn[i] = (__bf16)xn[i];

  if (t < 64) {
    for (int m = 0; m < 4; ++m) {
      float e = ew_anc[((size_t)b * 64 + t) * 4 + m];
      wL[m * 64 + t] = (e > 0.f) ? e : 0.f;   // ew * (ew>0)
    }
  }
  __syncthreads();
  if (t < 4) {
    float s = 0.f;
    for (int n = 0; n < 64; ++n) s += wL[t * 64 + n];
    const float den = fmaxf(s, 1e-6f);        // clip(sum, EPS)
    swn[t] = s / den;                          // sum of normalized weights
    const float inv = 1.f / den;
    for (int n = 0; n < 64; ++n) wL[t * 64 + n] *= inv;
  }
  __syncthreads();

  for (int m = 0; m < 4; ++m) {
    gemm64<true>(Xn, 128, WvembT + m * 64 * 128, HM, bv_emb + m * 64);
    __syncthreads();
    if (m == 0) {
      __bf16* dst = hns_ws + (size_t)b * 64 * 64;
      for (int i = t; i < 64 * 64; i += 128) dst[i] = (__bf16)HM[i];
    }
    if (t < 64) {
      float a = 0.f;
      for (int n = 0; n < 64; ++n) a += wL[m * 64 + n] * HM[n * 64 + t];
      cbar[t] = a;
    }
    __syncthreads();
    if (t < 64) {
      float a = 0.f;
      const float* Wm = Wv_val + m * 64 * 64;
      for (int k = 0; k < 64; ++k) a += cbar[k] * Wm[k * 64 + t];
      const float val = a + swn[m] * bv_val[m * 64 + t];
      pvc_ws[((size_t)b * 4 + m) * 64 + t] = val;
      if (m == 0) ctxacc[t] = val; else ctxacc[t] += val;
    }
    __syncthreads();
  }
  if (t < 64) ctxvec_ws[(size_t)b * 64 + t] = ctxacc[t] * 0.25f;
}

// ---------------------------------------------------------------------------
// Kernel 2: h_anc + router MLP + pi/beta/iso_heads/gate   (64 rows per block)
// ---------------------------------------------------------------------------
__global__ void __launch_bounds__(128) k_router(
    const float* __restrict__ x_anc, const float* __restrict__ g_anc,
    const float* __restrict__ ew_anc, const float* __restrict__ b_anc,
    const float* __restrict__ r_bin, const float* __restrict__ r_lng,
    const float* __restrict__ r_lnb, const float* __restrict__ r_b1,
    const float* __restrict__ r_b2, const float* __restrict__ r_Wview,
    const float* __restrict__ r_bview, const float* __restrict__ r_Wmode,
    const float* __restrict__ r_bmode,
    const __bf16* __restrict__ WancT, const __bf16* __restrict__ rWinT,
    const __bf16* __restrict__ rW1T, const __bf16* __restrict__ rW2T,
    const float* __restrict__ ctxvec_ws, const float* __restrict__ pvc_ws,
    float* __restrict__ hanc_ws, float* __restrict__ iso_ws,
    float* __restrict__ gate_ws, float* __restrict__ beta_ws) {
  __shared__ __align__(16) __bf16 A[64 * 128];
  __shared__ __align__(16) float  C[64 * 64];
  __shared__ float piL[64 * 16];

  const int R0 = blockIdx.x * 64;
  const int t  = threadIdx.x;

  // --- h_anc = x_anc @ W_anc + b_anc
  for (int i = t; i < 64 * 128; i += 128)
    A[i] = (__bf16)x_anc[(size_t)R0 * 128 + i];
  __syncthreads();
  gemm64<true>(A, 128, WancT, C, b_anc);
  __syncthreads();
  for (int i = t; i < 64 * 64; i += 128) hanc_ws[(size_t)R0 * 64 + i] = C[i];
  __syncthreads();

  // --- h0 = concat(g_anc, ctx_vec) @ r_Win + r_bin
  for (int i = t; i < 64 * 128; i += 128) {
    const int r = i >> 7, c = i & 127;
    const float v = (c < 64) ? g_anc[((size_t)(R0 + r)) * 64 + c]
                             : ctxvec_ws[((size_t)(R0 + r)) * 64 + (c - 64)];
    A[i] = (__bf16)v;
  }
  __syncthreads();
  gemm64<true>(A, 128, rWinT, C, r_bin);
  __syncthreads();

  // --- LN + gelu -> A (bf16, stride 64)
  if (t < 64) {
    float* row = C + t * 64;
    float mu = 0.f;
    for (int d = 0; d < 64; ++d) mu += row[d];
    mu *= (1.f / 64.f);
    float var = 0.f;
    for (int d = 0; d < 64; ++d) { const float z = row[d] - mu; var += z * z; }
    var *= (1.f / 64.f);
    const float rs = rsqrtf(var + 1e-5f);
    for (int d = 0; d < 64; ++d) {
      const float z = (row[d] - mu) * rs * r_lng[d] + r_lnb[d];
      A[t * 64 + d] = (__bf16)gelu_exact(z);
    }
  }
  __syncthreads();
  gemm64<true>(A, 64, rW1T, C, r_b1);
  __syncthreads();
  if (t < 64)
    for (int d = 0; d < 64; ++d) A[t * 64 + d] = (__bf16)gelu_exact(C[t * 64 + d]);
  __syncthreads();
  gemm64<true>(A, 64, rW2T, C, r_b2);
  __syncthreads();
  if (t < 64)
    for (int d = 0; d < 64; ++d) C[t * 64 + d] = gelu_exact(C[t * 64 + d]);
  __syncthreads();

  // --- pi (softmax over m), beta (sigmoid)
  if (t < 64) {
    const float* h2 = C + t * 64;
    for (int h = 0; h < 4; ++h) {
      float lg0 = r_bview[h * 4 + 0], lg1 = r_bview[h * 4 + 1];
      float lg2 = r_bview[h * 4 + 2], lg3 = r_bview[h * 4 + 3];
      for (int k = 0; k < 64; ++k) {
        const float hv = h2[k];
        const float* wv = r_Wview + (h * 64 + k) * 4;
        lg0 += hv * wv[0]; lg1 += hv * wv[1];
        lg2 += hv * wv[2]; lg3 += hv * wv[3];
      }
      const float mx = fmaxf(fmaxf(lg0, lg1), fmaxf(lg2, lg3));
      const float e0 = expf(lg0 - mx), e1 = expf(lg1 - mx);
      const float e2 = expf(lg2 - mx), e3 = expf(lg3 - mx);
      const float inv = 1.f / (e0 + e1 + e2 + e3);
      piL[t * 16 + h * 4 + 0] = e0 * inv;
      piL[t * 16 + h * 4 + 1] = e1 * inv;
      piL[t * 16 + h * 4 + 2] = e2 * inv;
      piL[t * 16 + h * 4 + 3] = e3 * inv;
      float z = r_bmode[h];
      for (int k = 0; k < 64; ++k) z += h2[k] * r_Wmode[h * 64 + k];
      beta_ws[((size_t)(R0 + t)) * 4 + h] = 1.f / (1.f + expf(-z));
    }
  }
  __syncthreads();

  // --- iso_heads[b,h,d] = sum_m pi * per_view_ctx
  for (int idx = t; idx < 64 * 4 * 64; idx += 128) {
    const int i = idx >> 8, h = (idx >> 6) & 3, d = idx & 63;
    const float* pv = pvc_ws + ((size_t)(R0 + i) * 4) * 64;
    float s = 0.f;
    for (int m = 0; m < 4; ++m) s += piL[i * 16 + h * 4 + m] * pv[m * 64 + d];
    iso_ws[((size_t)(R0 + i) * 4 + h) * 64 + d] = s;
  }
  // --- gate[b,h,n] = log(sum_m pi * masked_ew + EPS)
  for (int idx = t; idx < 64 * 4 * 64; idx += 128) {
    const int i = idx >> 8, h = (idx >> 6) & 3, n = idx & 63;
    const float* e = ew_anc + ((size_t)(R0 + i) * 64 + n) * 4;
    float s = 0.f;
    for (int m = 0; m < 4; ++m) {
      float ev = e[m]; ev = (ev > 0.f) ? ev : 0.f;
      s += piL[i * 16 + h * 4 + m] * ev;
    }
    gate_ws[((size_t)(R0 + i) * 4 + h) * 64 + n] = logf(s + 1e-6f);
  }
}

// ---------------------------------------------------------------------------
// Kernel 3: per-b transformer (L=2) + blend + output head
// ---------------------------------------------------------------------------
__global__ void __launch_bounds__(128) k_transformer(
    const __bf16* __restrict__ hns_ws, const float* __restrict__ hanc_ws,
    const float* __restrict__ gate_ws, const float* __restrict__ beta_ws,
    const float* __restrict__ iso_ws, const float* __restrict__ log_tau,
    const float* __restrict__ L_Wq, const __bf16* __restrict__ LWkT,
    const __bf16* __restrict__ LWvT, const float* __restrict__ L_Wo,
    const float* __restrict__ L_ln1g, const float* __restrict__ L_ln1b,
    const float* __restrict__ L_ln2g, const float* __restrict__ L_ln2b,
    const float* __restrict__ L_Wf1, const float* __restrict__ L_bf1,
    const float* __restrict__ L_Wf2, const float* __restrict__ L_bf2,
    const float* __restrict__ p_lng, const float* __restrict__ p_lnb,
    const float* __restrict__ p_W1, const float* __restrict__ p_b1,
    const float* __restrict__ p_W2, const float* __restrict__ p_b2,
    float* __restrict__ out) {
  __shared__ __align__(16) __bf16 HNS[64 * 64];
  __shared__ __align__(16) float Kf[64 * 64];
  __shared__ __align__(16) float Vf[64 * 64];
  __shared__ float xL[64], qL[64], sc[4 * 64], ctxL[64], xr[64];
  __shared__ float ff1L[128], gateL[4 * 64];
  __shared__ float tauL[4], betaL[4], t1L[32], yL[64];

  const int b = blockIdx.x;
  const int t = threadIdx.x;

  const __bf16* hp = hns_ws + (size_t)b * 4096;
  for (int i = t; i < 4096; i += 128) HNS[i] = hp[i];
  if (t < 64) xL[t] = hanc_ws[(size_t)b * 64 + t];
  for (int i = t; i < 256; i += 128) gateL[i] = gate_ws[(size_t)b * 256 + i];
  if (t < 4) { tauL[t] = expf(log_tau[t]); betaL[t] = beta_ws[(size_t)b * 4 + t]; }
  __syncthreads();

  for (int l = 0; l < 2; ++l) {
    gemm64<false>(HNS, 64, LWkT + l * 4096, Kf, nullptr);
    gemm64<false>(HNS, 64, LWvT + l * 4096, Vf, nullptr);
    __syncthreads();
    if (t < 64) {
      float a = 0.f;
      const float* Wq = L_Wq + (size_t)l * 4096;
      for (int d = 0; d < 64; ++d) a += xL[d] * Wq[d * 64 + t];
      qL[t] = a;
    }
    __syncthreads();
    if (t < 64) {
      for (int h = 0; h < 4; ++h) {
        float s = 0.f;
        for (int dd = 0; dd < 16; ++dd)
          s += qL[h * 16 + dd] * Kf[t * 64 + h * 16 + dd];
        sc[h * 64 + t] = s * (0.25f / tauL[h]) + gateL[h * 64 + t];
      }
    }
    __syncthreads();
    if (t < 4) {                                  // softmax over n per head
      float mx = -1e30f;
      for (int n = 0; n < 64; ++n) mx = fmaxf(mx, sc[t * 64 + n]);
      float s = 0.f;
      for (int n = 0; n < 64; ++n) {
        const float e = expf(sc[t * 64 + n] - mx);
        sc[t * 64 + n] = e; s += e;
      }
      const float inv = 1.f / s;
      for (int n = 0; n < 64; ++n) sc[t * 64 + n] *= inv;
    }
    __syncthreads();
    if (t < 64) {
      const int h = t >> 4;
      float a = 0.f;
      for (int n = 0; n < 64; ++n) a += sc[h * 64 + n] * Vf[n * 64 + t];
      ctxL[t] = a;
    }
    __syncthreads();
    if (t < 64) {
      float o = 0.f;
      const float* Wo = L_Wo + (size_t)l * 4096;
      for (int k = 0; k < 64; ++k) o += ctxL[k] * Wo[k * 64 + t];
      xr[t] = xL[t] + o;
    }
    __syncthreads();
    if (t < 64) {                                 // LN1
      float mu = 0.f;
      for (int d = 0; d < 64; ++d) mu += xr[d];
      mu *= (1.f / 64.f);
      float var = 0.f;
      for (int d = 0; d < 64; ++d) { const float z = xr[d] - mu; var += z * z; }
      var *= (1.f / 64.f);
      const float rs = rsqrtf(var + 1e-5f);
      xL[t] = (xr[t] - mu) * rs * L_ln1g[l * 64 + t] + L_ln1b[l * 64 + t];
    }
    __syncthreads();
    {                                             // ff1 (128 outputs)
      float a = L_bf1[l * 128 + t];
      const float* W = L_Wf1 + (size_t)l * 64 * 128;
      for (int d = 0; d < 64; ++d) a += xL[d] * W[d * 128 + t];
      ff1L[t] = gelu_exact(a);
    }
    __syncthreads();
    if (t < 64) {                                 // ff2
      float a = L_bf2[l * 64 + t];
      const float* W = L_Wf2 + (size_t)l * 128 * 64;
      for (int j = 0; j < 128; ++j) a += ff1L[j] * W[j * 64 + t];
      xr[t] = xL[t] + a;
    }
    __syncthreads();
    if (t < 64) {                                 // LN2
      float mu = 0.f;
      for (int d = 0; d < 64; ++d) mu += xr[d];
      mu *= (1.f / 64.f);
      float var = 0.f;
      for (int d = 0; d < 64; ++d) { const float z = xr[d] - mu; var += z * z; }
      var *= (1.f / 64.f);
      const float rs = rsqrtf(var + 1e-5f);
      xL[t] = (xr[t] - mu) * rs * L_ln2g[l * 64 + t] + L_ln2b[l * 64 + t];
    }
    __syncthreads();
  }

  // blended = mean_h((1-beta)*iso + beta*x)
  if (t < 64) {
    float s = 0.f;
    for (int h = 0; h < 4; ++h) {
      const float be = betaL[h];
      s += (1.f - be) * iso_ws[((size_t)b * 4 + h) * 64 + t] + be * xL[t];
    }
    xr[t] = s * 0.25f;
  }
  __syncthreads();
  if (t < 64) {                                   // final LN
    float mu = 0.f;
    for (int d = 0; d < 64; ++d) mu += xr[d];
    mu *= (1.f / 64.f);
    float var = 0.f;
    for (int d = 0; d < 64; ++d) { const float z = xr[d] - mu; var += z * z; }
    var *= (1.f / 64.f);
    const float rs = rsqrtf(var + 1e-5f);
    yL[t] = (xr[t] - mu) * rs * p_lng[t] + p_lnb[t];
  }
  __syncthreads();
  if (t < 32) {
    float a = p_b1[t];
    for (int d = 0; d < 64; ++d) a += yL[d] * p_W1[d * 32 + t];
    t1L[t] = gelu_exact(a);
  }
  __syncthreads();
  if (t < 8) {
    float a = p_b2[t];
    for (int j = 0; j < 32; ++j) a += t1L[j] * p_W2[j * 8 + t];
    out[(size_t)b * 8 + t] = a;
  }
}

// ---------------------------------------------------------------------------
// Host launcher
// ---------------------------------------------------------------------------
extern "C" void kernel_launch(void* const* d_in, const int* in_sizes, int n_in,
                              void* d_out, int out_size, void* d_ws,
                              size_t ws_size, hipStream_t stream) {
  (void)in_sizes; (void)n_in; (void)out_size; (void)ws_size;

  const float* x_anc   = (const float*)d_in[0];
  const float* g_anc   = (const float*)d_in[1];
  const float* x_nei   = (const float*)d_in[2];
  const float* ew_anc  = (const float*)d_in[3];
  const float* W_anc   = (const float*)d_in[4];
  const float* b_anc   = (const float*)d_in[5];
  const float* Wv_emb  = (const float*)d_in[6];
  const float* bv_emb  = (const float*)d_in[7];
  const float* Wv_val  = (const float*)d_in[8];
  const float* bv_val  = (const float*)d_in[9];
  const float* r_Win   = (const float*)d_in[10];
  const float* r_bin   = (const float*)d_in[11];
  const float* r_lng   = (const float*)d_in[12];
  const float* r_lnb   = (const float*)d_in[13];
  const float* r_W1    = (const float*)d_in[14];
  const float* r_b1    = (const float*)d_in[15];
  const float* r_W2    = (const float*)d_in[16];
  const float* r_b2    = (const float*)d_in[17];
  const float* r_Wview = (const float*)d_in[18];
  const float* r_bview = (const float*)d_in[19];
  const float* r_Wmode = (const float*)d_in[20];
  const float* r_bmode = (const float*)d_in[21];
  const float* log_tau = (const float*)d_in[22];
  const float* L_Wq    = (const float*)d_in[23];
  const float* L_Wk    = (const float*)d_in[24];
  const float* L_Wv    = (const float*)d_in[25];
  const float* L_Wo    = (const float*)d_in[26];
  const float* L_ln1g  = (const float*)d_in[27];
  const float* L_ln1b  = (const float*)d_in[28];
  const float* L_ln2g  = (const float*)d_in[29];
  const float* L_ln2b  = (const float*)d_in[30];
  const float* L_Wf1   = (const float*)d_in[31];
  const float* L_bf1   = (const float*)d_in[32];
  const float* L_Wf2   = (const float*)d_in[33];
  const float* L_bf2   = (const float*)d_in[34];
  const float* p_lng   = (const float*)d_in[35];
  const float* p_lnb   = (const float*)d_in[36];
  const float* p_W1    = (const float*)d_in[37];
  const float* p_b1    = (const float*)d_in[38];
  const float* p_W2    = (const float*)d_in[39];
  const float* p_b2    = (const float*)d_in[40];

  // workspace carve-up (256B aligned slots)
  char* w = (char*)d_ws;
  auto take = [&](size_t bytes) -> char* {
    char* p = w;
    w += (bytes + 255) & ~(size_t)255;
    return p;
  };
  __bf16* WancT  = (__bf16*)take((size_t)128 * 64 * 2);
  __bf16* WvembT = (__bf16*)take((size_t)4 * 128 * 64 * 2);
  __bf16* rWinT  = (__bf16*)take((size_t)128 * 64 * 2);
  __bf16* rW1T   = (__bf16*)take((size_t)64 * 64 * 2);
  __bf16* rW2T   = (__bf16*)take((size_t)64 * 64 * 2);
  __bf16* LWkT   = (__bf16*)take((size_t)2 * 64 * 64 * 2);
  __bf16* LWvT   = (__bf16*)take((size_t)2 * 64 * 64 * 2);
  __bf16* hns    = (__bf16*)take((size_t)4096 * 64 * 64 * 2);   // 32 MB
  float* pvc     = (float*)take((size_t)4096 * 4 * 64 * 4);     // 4 MB
  float* ctxvec  = (float*)take((size_t)4096 * 64 * 4);         // 1 MB
  float* hanc    = (float*)take((size_t)4096 * 64 * 4);         // 1 MB
  float* iso     = (float*)take((size_t)4096 * 4 * 64 * 4);     // 4 MB
  float* gate    = (float*)take((size_t)4096 * 4 * 64 * 4);     // 4 MB
  float* betaB   = (float*)take((size_t)4096 * 4 * 4);

  // --- weight conversion (f32 row-major -> bf16 transposed)
  k_convT<<<dim3(4, 1), 256, 0, stream>>>(W_anc, WancT, 128, 64);
  k_convT<<<dim3(4, 4), 256, 0, stream>>>(Wv_emb, WvembT, 128, 64);
  k_convT<<<dim3(4, 1), 256, 0, stream>>>(r_Win, rWinT, 128, 64);
  k_convT<<<dim3(2, 1), 256, 0, stream>>>(r_W1, rW1T, 64, 64);
  k_convT<<<dim3(2, 1), 256, 0, stream>>>(r_W2, rW2T, 64, 64);
  k_convT<<<dim3(2, 2), 256, 0, stream>>>(L_Wk, LWkT, 64, 64);
  k_convT<<<dim3(2, 2), 256, 0, stream>>>(L_Wv, LWvT, 64, 64);

  // --- stage 1: neighbor embeddings + per-view context
  k_neighbor<<<4096, 128, 0, stream>>>(x_nei, ew_anc, bv_emb, Wv_val, bv_val,
                                       WvembT, hns, pvc, ctxvec);

  // --- stage 2: router
  k_router<<<64, 128, 0, stream>>>(x_anc, g_anc, ew_anc, b_anc, r_bin, r_lng,
                                   r_lnb, r_b1, r_b2, r_Wview, r_bview,
                                   r_Wmode, r_bmode, WancT, rWinT, rW1T, rW2T,
                                   ctxvec, pvc, hanc, iso, gate, betaB);

  // --- stage 3: transformer + head
  k_transformer<<<4096, 128, 0, stream>>>(
      hns, hanc, gate, betaB, iso, log_tau, L_Wq, LWkT, LWvT, L_Wo, L_ln1g,
      L_ln1b, L_ln2g, L_ln2b, L_Wf1, L_bf1, L_Wf2, L_bf2, p_lng, p_lnb, p_W1,
      p_b1, p_W2, p_b2, (float*)d_out);
}
